// MoE_14439680049329
// MI455X (gfx1250) — matrix operand order repaired and checked by reading, not compiled, not run on hardware
//
#include <hip/hip_runtime.h>
#include <math.h>

typedef __attribute__((ext_vector_type(16))) __bf16 bf16x16;
typedef __attribute__((ext_vector_type(8)))  float  f32x8;
typedef int v4i_vec __attribute__((vector_size(16)));   // matches async builtin param
typedef unsigned int u32x4 __attribute__((ext_vector_type(4)));
typedef int i32x8 __attribute__((ext_vector_type(8)));
typedef int i32x4 __attribute__((ext_vector_type(4)));

#define E_EXP 4
#define BATCH 64
#define CIN   256
#define BN    64
#define LLEN  4096
#define TL    128         // L-tile per workgroup (8 waves x 16 cols)
#define NTHR  256

// packed-weight layout (ushort units), per expert, all mt-major:
//  W1: 4mt x 8kt = 32 frags ; W2: 3taps x (4mt x 2kt) = 24
//  W3: 16mt x 2kt = 32      ; Wp: 16mt x 8kt = 128
#define FRAG_USH 512                       // 32 lanes * 16 halfs = 1 KB
#define W1OFF 0
#define W2OFF (32 * FRAG_USH)
#define W3OFF (56 * FRAG_USH)
#define WPOFF (88 * FRAG_USH)
#define PEREXP (216 * FRAG_USH)            // 110592 ushorts per expert

static __device__ __forceinline__ unsigned short f2bf(float f) {
    unsigned int u = __float_as_uint(f);
    return (unsigned short)((u + 0x7FFFu + ((u >> 16) & 1u)) >> 16);
}
static __device__ __forceinline__ float bf2f(unsigned short h) {
    return __uint_as_float(((unsigned int)h) << 16);
}

// load a 32B/lane fragment (LDS: two ds_load_b128)
static __device__ __forceinline__ bf16x16 ldfrag32B(const unsigned short* p) {
    union { bf16x16 v; uint4 q[2]; } u;
    u.q[0] = ((const uint4*)p)[0];
    u.q[1] = ((const uint4*)p)[1];
    return u.v;
}

// ---- staging path A: per-lane async-to-LDS (ASYNCcnt) ---------------------
static __device__ __forceinline__ void stage_async(const unsigned short* __restrict__ g,
                                                   unsigned short* l, int nbytes, int tid) {
#if __has_builtin(__builtin_amdgcn_global_load_async_to_lds_b128)
    for (int off = tid * 16; off < nbytes; off += NTHR * 16) {
        __builtin_amdgcn_global_load_async_to_lds_b128(
            (v4i_vec*)((char*)(const_cast<unsigned short*>(g)) + off),
            (v4i_vec*)((char*)l + off), 0, 0);
    }
#else
    for (int off = tid * 16; off < nbytes; off += NTHR * 16)
        *(uint4*)((char*)l + off) = *(const uint4*)((const char*)g + off);
#endif
}
static __device__ __forceinline__ void async_wait_bar() {
#if __has_builtin(__builtin_amdgcn_global_load_async_to_lds_b128)
#if __has_builtin(__builtin_amdgcn_s_wait_asynccnt)
    __builtin_amdgcn_s_wait_asynccnt(0);
#else
    asm volatile("s_wait_asynccnt 0" ::: "memory");
#endif
#endif
    __syncthreads();
}

// ---- staging path B: Tensor Data Mover (TENSORcnt) ------------------------
// Flat 1-D copy: data_size=8B, tile_dim0=tensor_dim0=nbytes/8, D# type=2.
// Issued by wave 0 only (TDM is wave-level and ignores EXEC).
static __device__ __forceinline__ void stage_tdm(const unsigned short* __restrict__ g,
                                                 unsigned short* l, int nbytes, int tid) {
#if __has_builtin(__builtin_amdgcn_tensor_load_to_lds)
    if (tid < 32) {
        unsigned long long ga = (unsigned long long)(size_t)g;
        unsigned int la = (unsigned int)(size_t)l;       // low 32 bits = LDS offset
        int n8 = nbytes >> 3;
        u32x4 g0;
        g0[0] = 1u;                                      // count=1, user D#
        g0[1] = la;                                      // lds_addr
        g0[2] = (unsigned int)ga;                        // global_addr[31:0]
        g0[3] = (unsigned int)((ga >> 32) & 0x01FFFFFFu) | 0x80000000u;  // type=2
        i32x8 g1;
        g1[0] = 0x00030000;                              // data_size=8B, no multicast
        g1[1] = (n8 & 0xFFFF) << 16;                     // tensor_dim0[15:0]
        g1[2] = (n8 >> 16) & 0xFFFF;                     // tensor_dim0[31:16], dim1=0
        g1[3] = (n8 & 0xFFFF) << 16;                     // tile_dim0
        g1[4] = 0;
        g1[5] = n8;                                      // tensor_dim0_stride
        g1[6] = 0;
        g1[7] = 0;
        i32x4 z4 = {0, 0, 0, 0};
        i32x8 z8 = {0, 0, 0, 0, 0, 0, 0, 0};
        __builtin_amdgcn_tensor_load_to_lds(g0, g1, z4, z4, z8, 0);
    }
#else
    stage_async(g, l, nbytes, tid);
#endif
}
static __device__ __forceinline__ void tdm_wait_bar(int tid) {
#if __has_builtin(__builtin_amdgcn_tensor_load_to_lds)
#if __has_builtin(__builtin_amdgcn_s_wait_tensorcnt)
    if (tid < 32) __builtin_amdgcn_s_wait_tensorcnt(0);
#else
    if (tid < 32) asm volatile("s_wait_tensorcnt 0" ::: "memory");
#endif
    __syncthreads();
#else
    async_wait_bar();
#endif
}

// ---------------------------------------------------------------------------
// Pass 0: pack fp32 weights -> bf16 WMMA A-fragments (contiguous-K 16x32 layout)
// row = lane&15; lane<16: K={0..15}; lane>=16: K={16..31}
// ---------------------------------------------------------------------------
__global__ void moe_pack_kernel(const float* __restrict__ src,
                                unsigned short* __restrict__ dst,
                                int MT, int KT, int kmajor,
                                int rstride, int cstride,
                                long src_estride, long dst_estride_ush,
                                int src_tstride, int dst_tstride_frags) {
    const int lane = threadIdx.x;
    const int f = blockIdx.x;
    const int e = blockIdx.y;
    const int t = blockIdx.z;
    int mt, kt;
    if (kmajor) { kt = f / MT; mt = f - kt * MT; }
    else        { mt = f / KT; kt = f - mt * KT; }
    const float* s = src + (long)e * src_estride + (long)t * src_tstride;
    unsigned short* d = dst + (long)e * dst_estride_ush +
                        ((long)t * dst_tstride_frags + f) * FRAG_USH + lane * 16;
    const int row = 16 * mt + (lane & 15);
    const int kb = (lane < 16) ? 0 : 16;
    union { unsigned short h[16]; uint4 q[2]; } u;
#pragma unroll
    for (int j = 0; j < 16; ++j) {
        int k = 32 * kt + kb + j;
        u.h[j] = f2bf(s[(long)row * rstride + (long)k * cstride]);
    }
    ((uint4*)d)[0] = u.q[0];
    ((uint4*)d)[1] = u.q[1];
}

// ---------------------------------------------------------------------------
// Pass 1: gate_x[b][c] = mean_l x[b,c,l]
// ---------------------------------------------------------------------------
__global__ void moe_mean_kernel(const float* __restrict__ x, float* __restrict__ gate_x) {
    const int bc = blockIdx.x;
    const float* p = x + (long)bc * LLEN;
    float s = 0.f;
    for (int i = threadIdx.x; i < LLEN; i += 256) s += p[i];
    __shared__ float red[256];
    red[threadIdx.x] = s;
    __syncthreads();
    for (int off = 128; off > 0; off >>= 1) {
        if ((int)threadIdx.x < off) red[threadIdx.x] += red[threadIdx.x + off];
        __syncthreads();
    }
    if (threadIdx.x == 0) gate_x[bc] = red[0] * (1.f / (float)LLEN);
}

// ---------------------------------------------------------------------------
// Pass 2: noisy top-k gating + aux loss (single block, 64 threads = 1 per b)
// ---------------------------------------------------------------------------
static __device__ __forceinline__ float softplus_f(float v) {
    return (v > 20.f) ? v : log1pf(expf(v));
}
static __device__ __forceinline__ float ncdf(float z) {
    return 0.5f * erfcf(-z * 0.70710678118654752f);
}
static __device__ __forceinline__ float cv2(const float* v) {
    float m = 0.25f * (v[0] + v[1] + v[2] + v[3]);
    float var = 0.f;
    for (int i = 0; i < 4; ++i) { float d = v[i] - m; var += d * d; }
    var *= (1.f / 3.f);                       // ddof = 1
    return var / (m * m + 1e-10f);
}

__global__ void moe_gate_kernel(const float* __restrict__ gate_x,
                                const float* __restrict__ noise,
                                const float* __restrict__ w_gate,
                                const float* __restrict__ w_noise,
                                float* __restrict__ gates,
                                float* __restrict__ loss_out) {
    const int b = threadIdx.x;            // 0..63
    float clean[4] = {0, 0, 0, 0}, rawn[4] = {0, 0, 0, 0};
    const float* gx = gate_x + b * CIN;
    for (int c = 0; c < CIN; ++c) {
        float v = gx[c];
#pragma unroll
        for (int e = 0; e < 4; ++e) {
            clean[e] = fmaf(v, w_gate[c * 4 + e], clean[e]);
            rawn[e]  = fmaf(v, w_noise[c * 4 + e], rawn[e]);
        }
    }
    float stdv[4], noisy[4];
#pragma unroll
    for (int e = 0; e < 4; ++e) {
        stdv[e]  = softplus_f(rawn[e]) + 0.01f;
        noisy[e] = clean[e] + noise[b * 4 + e] * stdv[e];
    }
    float mx = fmaxf(fmaxf(noisy[0], noisy[1]), fmaxf(noisy[2], noisy[3]));
    float ex[4], se = 0.f;
#pragma unroll
    for (int e = 0; e < 4; ++e) { ex[e] = expf(noisy[e] - mx); se += ex[e]; }
    float v[4]; int id[4];
#pragma unroll
    for (int e = 0; e < 4; ++e) { v[e] = ex[e] / se; id[e] = e; }
#define CSWAP(i, j)                                                            \
    if (v[j] > v[i]) {                                                         \
        float tv = v[i]; v[i] = v[j]; v[j] = tv;                               \
        int ti = id[i]; id[i] = id[j]; id[j] = ti;                             \
    }
    CSWAP(0, 1) CSWAP(2, 3) CSWAP(0, 2) CSWAP(1, 3) CSWAP(1, 2)
#undef CSWAP
    float m2 = fmaxf(v[0], v[1]);
    float e0 = expf(v[0] - m2), e1 = expf(v[1] - m2);
    float g0 = e0 / (e0 + e1), g1 = e1 / (e0 + e1);
    float grow[4] = {0, 0, 0, 0};
    grow[id[0]] = g0;
    grow[id[1]] = g1;
#pragma unroll
    for (int e = 0; e < 4; ++e) gates[b * 4 + e] = grow[e];

    const float thr_in = v[2], thr_out = v[1];
    float loadrow[4];
#pragma unroll
    for (int e = 0; e < 4; ++e) {
        bool is_in = noisy[e] > thr_in;
        float pin  = ncdf((clean[e] - thr_in)  / stdv[e]);
        float pout = ncdf((clean[e] - thr_out) / stdv[e]);
        loadrow[e] = is_in ? pin : pout;
    }
    __shared__ float simp[4][64], sld[4][64];
#pragma unroll
    for (int e = 0; e < 4; ++e) { simp[e][b] = grow[e]; sld[e][b] = loadrow[e]; }
    __syncthreads();
    if (b == 0) {
        float imp[4], ld[4];
#pragma unroll
        for (int e = 0; e < 4; ++e) {
            float si = 0.f, sl = 0.f;
            for (int i = 0; i < 64; ++i) { si += simp[e][i]; sl += sld[e][i]; }
            imp[e] = si; ld[e] = sl;
        }
        loss_out[0] = 0.01f * (cv2(imp) + cv2(ld));
    }
}

// ---------------------------------------------------------------------------
// Pass 3: fused expert pipeline. 8 waves, TL=128; weights staged in LDS once
// per block (W1/W2 via async-to-LDS, Wp/W3 chunks via TDM) shared by all
// waves; B fragments hoisted (1 load per 4 WMMAs); VGPRs capped at 256.
// ---------------------------------------------------------------------------
__global__ __launch_bounds__(NTHR, 4)
void moe_main_kernel(const float* __restrict__ x,
                     const float* __restrict__ gates,
                     const unsigned short* __restrict__ pk,
                     const float* __restrict__ W1,
                     const float* __restrict__ b1,
                     const float* __restrict__ b2,
                     const float* __restrict__ b3,
                     const float* __restrict__ bp,
                     float* __restrict__ y) {
    __shared__ __align__(16) unsigned short xs[(TL + 2) * CIN];   // 66.5 KB
    __shared__ __align__(16) unsigned short h1s[(TL + 2) * BN];   // 16.6 KB
    __shared__ __align__(16) unsigned short h2s[TL * BN];         // 16.0 KB
    __shared__ __align__(16) unsigned short wbuf[40 * FRAG_USH];  // 40.0 KB

    const int tid  = threadIdx.x;
    const int b    = blockIdx.y;
    const int l0   = blockIdx.x * TL;
    const int lane = tid & 31;
    const int n0   = (tid >> 5) * 16;                  // wave's column base
    const int lc   = lane & 15;
    const int rofs = (lane < 16) ? 0 : 8;              // C/D row offset
    const int kbB  = (lane < 16) ? 0 : 16;             // B fragment K base

    // ---- stage x tile into LDS as bf16, transposed (K=c contiguous) ----
    const float* xb = x + (long)b * CIN * LLEN;
    for (int idx = tid; idx < (TL + 2) * CIN; idx += NTHR) {
        int c = idx / (TL + 2);
        int col = idx - c * (TL + 2);
        int gl = l0 - 1 + col;
        float val = (gl >= 0 && gl < LLEN) ? xb[(long)c * LLEN + gl] : 0.f;
        xs[col * CIN + c] = f2bf(val);
    }
    __syncthreads();

    float yacc[16][8];
#pragma unroll
    for (int m = 0; m < 16; ++m)
#pragma unroll
        for (int r = 0; r < 8; ++r) yacc[m][r] = 0.f;

    for (int e = 0; e < E_EXP; ++e) {
        const float g = gates[b * 4 + e];
        if (g == 0.f) continue;                         // uniform per block
        const unsigned short* pe = pk + (long)e * PEREXP;
        const float* b1e = b1 + e * BN;
        const float* b2e = b2 + e * BN;
        const float* b3e = b3 + e * CIN;
        const float* bpe = bp + e * CIN;

        __syncthreads();                 // wbuf + h1s free from previous expert
        stage_async(pe + W1OFF, wbuf, 32 * FRAG_USH * 2, tid);
        // halo columns of h1 (zero-padded conv2 boundary) via VALU, overlapped
        if (tid < 128) {
            int oc = tid & 63, side = tid >> 6;
            int col = side ? (TL + 1) : 0;
            int gl = l0 - 1 + col;
            float h = 0.f;
            if (gl >= 0 && gl < LLEN) {
                float s = b1e[oc];
                const float* w = W1 + ((long)e * BN + oc) * CIN;
                for (int c = 0; c < CIN; ++c) s = fmaf(w[c], bf2f(xs[col * CIN + c]), s);
                h = fmaxf(s, 0.f);
            }
            h1s[col * BN + oc] = f2bf(h);
        }
        async_wait_bar();

        // -------- stage 1: h1 = relu(W1 @ x + b1), 64 rows ----------------
        {
            f32x8 acc1[4];
#pragma unroll
            for (int mt = 0; mt < 4; ++mt)
#pragma unroll
                for (int r = 0; r < 8; ++r) acc1[mt][r] = b1e[16 * mt + rofs + r];
#pragma unroll
            for (int kt = 0; kt < 8; ++kt) {
                bf16x16 bm = ldfrag32B(&xs[(1 + n0 + lc) * CIN + kt * 32 + kbB]);
#pragma unroll
                for (int mt = 0; mt < 4; ++mt) {
                    bf16x16 a = ldfrag32B(wbuf + ((mt * 8 + kt) * 32 + lane) * 16);
                    acc1[mt] = __builtin_amdgcn_wmma_f32_16x16x32_bf16(
                        false, a, false, bm, (short)0, acc1[mt], false, false);
                }
            }
#pragma unroll
            for (int mt = 0; mt < 4; ++mt) {
                union { unsigned short h[8]; uint4 q; } t;
#pragma unroll
                for (int r = 0; r < 8; ++r) t.h[r] = f2bf(fmaxf(acc1[mt][r], 0.f));
                *(uint4*)&h1s[(1 + n0 + lc) * BN + 16 * mt + rofs] = t.q;
            }
        }
        __syncthreads();                 // h1s ready; wbuf reads done
        stage_async(pe + W2OFF, wbuf, 24 * FRAG_USH * 2, tid);
        async_wait_bar();

        // -------- stage 2: h2 = relu(conv3(h1) + b2) ----------------------
        {
            f32x8 acc2[4];
#pragma unroll
            for (int mt = 0; mt < 4; ++mt)
#pragma unroll
                for (int r = 0; r < 8; ++r) acc2[mt][r] = b2e[16 * mt + rofs + r];
#pragma unroll
            for (int tap = 0; tap < 3; ++tap)
#pragma unroll
                for (int kt = 0; kt < 2; ++kt) {
                    bf16x16 bm = ldfrag32B(&h1s[(n0 + lc + tap) * BN + kt * 32 + kbB]);
#pragma unroll
                    for (int mt = 0; mt < 4; ++mt) {
                        bf16x16 a = ldfrag32B(wbuf + ((tap * 8 + mt * 2 + kt) * 32 + lane) * 16);
                        acc2[mt] = __builtin_amdgcn_wmma_f32_16x16x32_bf16(
                            false, a, false, bm, (short)0, acc2[mt], false, false);
                    }
                }
#pragma unroll
            for (int mt = 0; mt < 4; ++mt) {
                union { unsigned short h[8]; uint4 q; } t;
#pragma unroll
                for (int r = 0; r < 8; ++r) t.h[r] = f2bf(fmaxf(acc2[mt][r], 0.f));
                *(uint4*)&h2s[(n0 + lc) * BN + 16 * mt + rofs] = t.q;
            }
        }
        __syncthreads();                 // h2s ready; wbuf reads done

        // -------- stage 3, chunked by mt (4 rows-of-16 per chunk):
        //          acc = Wp@x + W3@h2 + b3 + bp ; y += g*relu(acc) ----------
#pragma unroll
        for (int c = 0; c < 4; ++c) {
            stage_tdm(pe + WPOFF + c * 32 * FRAG_USH, wbuf, 32 * FRAG_USH * 2, tid);
            stage_tdm(pe + W3OFF + c * 8 * FRAG_USH, wbuf + 32 * FRAG_USH,
                      8 * FRAG_USH * 2, tid);
            tdm_wait_bar(tid);
            f32x8 acc3[4];
#pragma unroll
            for (int ml = 0; ml < 4; ++ml)
#pragma unroll
                for (int r = 0; r < 8; ++r) {
                    int row = 16 * (4 * c + ml) + rofs + r;
                    acc3[ml][r] = b3e[row] + bpe[row];
                }
#pragma unroll
            for (int kt = 0; kt < 8; ++kt) {            // Wp @ x
                bf16x16 bm = ldfrag32B(&xs[(1 + n0 + lc) * CIN + kt * 32 + kbB]);
#pragma unroll
                for (int ml = 0; ml < 4; ++ml) {
                    bf16x16 a = ldfrag32B(wbuf + ((ml * 8 + kt) * 32 + lane) * 16);
                    acc3[ml] = __builtin_amdgcn_wmma_f32_16x16x32_bf16(
                        false, a, false, bm, (short)0, acc3[ml], false, false);
                }
            }
#pragma unroll
            for (int kt = 0; kt < 2; ++kt) {            // W3 @ h2
                bf16x16 bm = ldfrag32B(&h2s[(n0 + lc) * BN + kt * 32 + kbB]);
#pragma unroll
                for (int ml = 0; ml < 4; ++ml) {
                    bf16x16 a = ldfrag32B(wbuf + ((32 + ml * 2 + kt) * 32 + lane) * 16);
                    acc3[ml] = __builtin_amdgcn_wmma_f32_16x16x32_bf16(
                        false, a, false, bm, (short)0, acc3[ml], false, false);
                }
            }
#pragma unroll
            for (int ml = 0; ml < 4; ++ml)
#pragma unroll
                for (int r = 0; r < 8; ++r)
                    yacc[4 * c + ml][r] += g * fmaxf(acc3[ml][r], 0.f);
            __syncthreads();             // wbuf reads done before next chunk
        }
        // top-of-loop barrier protects wbuf/h1s for the next expert
    }

    // -------- write y tile ------------------------------------------------
    float* yb = y + (long)b * CIN * LLEN + l0 + n0 + lc;
#pragma unroll
    for (int mt = 0; mt < 16; ++mt)
#pragma unroll
        for (int r = 0; r < 8; ++r)
            yb[(long)(16 * mt + rofs + r) * LLEN] = yacc[mt][r];
}

// ---------------------------------------------------------------------------
extern "C" void kernel_launch(void* const* d_in, const int* in_sizes, int n_in,
                              void* d_out, int out_size, void* d_ws, size_t ws_size,
                              hipStream_t stream) {
    (void)in_sizes; (void)n_in; (void)out_size; (void)ws_size;
    const float* x       = (const float*)d_in[0];
    const float* noise   = (const float*)d_in[1];
    const float* w_gate  = (const float*)d_in[2];
    const float* w_noise = (const float*)d_in[3];
    const float* W1      = (const float*)d_in[4];
    const float* b1      = (const float*)d_in[5];
    const float* W2      = (const float*)d_in[6];
    const float* b2      = (const float*)d_in[7];
    const float* W3      = (const float*)d_in[8];
    const float* b3      = (const float*)d_in[9];
    const float* Wp      = (const float*)d_in[10];
    const float* bp      = (const float*)d_in[11];

    float* y = (float*)d_out;
    float* loss_out = y + (long)BATCH * CIN * LLEN;

    char* ws = (char*)d_ws;
    float* gate_x = (float*)ws;                                  // 16384 f32
    float* gates  = (float*)(ws + 65536);                        // 256 f32
    unsigned short* pk = (unsigned short*)(ws + 66560);          // packed bf16 frags

    // pack weights into bf16 WMMA A fragments (all mt-major)
    moe_pack_kernel<<<dim3(32, 4, 1), 32, 0, stream>>>(
        W1, pk + W1OFF, 4, 8, 0, CIN, 1, (long)BN * CIN, PEREXP, 0, 0);
    moe_pack_kernel<<<dim3(8, 4, 3), 32, 0, stream>>>(
        W2, pk + W2OFF, 4, 2, 0, BN * 3, 3, (long)BN * BN * 3, PEREXP, 1, 8);
    moe_pack_kernel<<<dim3(32, 4, 1), 32, 0, stream>>>(
        W3, pk + W3OFF, 16, 2, 0, BN, 1, (long)CIN * BN, PEREXP, 0, 0);
    moe_pack_kernel<<<dim3(128, 4, 1), 32, 0, stream>>>(
        Wp, pk + WPOFF, 16, 8, 0, CIN, 1, (long)CIN * CIN, PEREXP, 0, 0);

    // gating
    moe_mean_kernel<<<dim3(BATCH * CIN), 256, 0, stream>>>(x, gate_x);
    moe_gate_kernel<<<dim3(1), 64, 0, stream>>>(gate_x, noise, w_gate, w_noise,
                                                gates, loss_out);

    // fused experts
    moe_main_kernel<<<dim3(LLEN / TL, BATCH), NTHR, 0, stream>>>(
        x, gates, pk, W1, b1, b2, b3, bp, y);
}
